// Block_4080218931561
// MI455X (gfx1250) — matrix operand order repaired and checked by reading, not compiled, not run on hardware
//
#include <hip/hip_runtime.h>
#include <cstdint>
#include <cstddef>

// ---------------- types ----------------
typedef __attribute__((ext_vector_type(16))) __bf16 v16bf;
typedef __attribute__((ext_vector_type(8)))  float  v8f;

union Frag16 { v16bf v; uint4 q[2]; };

static __device__ __forceinline__ uint4 ld16g(const __bf16* p) {
  return *reinterpret_cast<const uint4*>(p);
}

static __device__ __forceinline__ v8f wmma_bf16(v16bf a, v16bf b, v8f c) {
  return __builtin_amdgcn_wmma_f32_16x16x32_bf16(false, a, false, b, (short)0, c, false, false);
}

// ---------------- problem constants ----------------
constexpr int N_  = 16384;
constexpr int C_  = 512;
constexpr int H_  = 8;
constexpr int K_  = 1024;
constexpr int P_  = 16;     // N_/K_
constexpr float SCALE_ = 0.125f;   // (C/H)^-0.5 = 64^-0.5
constexpr float EPS_ = 1e-5f;

// ---------------- workspace layout (bytes) ----------------
constexpr size_t SZ_CPEW  = 27ull*512*512*2;
constexpr size_t SZ_CC    = 512ull*512*2;
constexpr size_t SZ_QKVW  = 3ull*512*512*2;
constexpr size_t SZ_FCW   = 2048ull*512*2;
constexpr size_t OFF_CPEW   = 0;
constexpr size_t OFF_CPELIN = OFF_CPEW + SZ_CPEW;
constexpr size_t OFF_QKVW   = OFF_CPELIN + SZ_CC;
constexpr size_t OFF_PROJW  = OFF_QKVW + SZ_QKVW;
constexpr size_t OFF_FC1W   = OFF_PROJW + SZ_CC;
constexpr size_t OFF_FC2W   = OFF_FC1W + SZ_FCW;
constexpr size_t SZ_ACT1    = (size_t)N_*C_*2;        // bf16 [N,C]
constexpr size_t OFF_ACT1   = OFF_FC2W + SZ_FCW;
constexpr size_t SZ_ACT4    = (size_t)N_*4*C_*2;      // bf16 [N,4C] (also q/k/vT)
constexpr size_t OFF_ACT4   = OFF_ACT1 + SZ_ACT1;
constexpr size_t SZ_F       = (size_t)N_*C_*4;        // f32 [N,C]
constexpr size_t OFF_TMPF   = OFF_ACT4 + SZ_ACT4;
constexpr size_t OFF_FEATR  = OFF_TMPF + SZ_F;

// ---------------- elementwise helpers ----------------
__global__ void f2bf_kernel(const float* __restrict__ in, __bf16* __restrict__ out, int n) {
  int i = blockIdx.x * blockDim.x + threadIdx.x;
  if (i < n) out[i] = (__bf16)in[i];
}

__global__ void add_kernel(const float4* __restrict__ a, const float4* __restrict__ b,
                           float4* __restrict__ out, int n4) {
  int i = blockIdx.x * blockDim.x + threadIdx.x;
  if (i < n4) {
    float4 x = a[i], y = b[i];
    out[i] = make_float4(x.x + y.x, x.y + y.y, x.z + y.z, x.w + y.w);
  }
}

// o_final[order[j]] = bf16(o_serial[j])
__global__ void scatter_kernel(const float* __restrict__ in, const int* __restrict__ order,
                               __bf16* __restrict__ out) {
  int i = blockIdx.x * blockDim.x + threadIdx.x;   // i over N*C exactly
  int j = i >> 9, c = i & 511;
  out[(size_t)order[j] * C_ + c] = (__bf16)in[i];
}

// ---------------- LayerNorm (C = 512, 128 threads x 4 elems) ----------------
// MODE 0: outF = base + LN(x);  MODE 1: outB = bf16(LN(x))
template<int MODE>
__global__ __launch_bounds__(128) void ln_kernel(const float* __restrict__ x,
                                                 const float* __restrict__ g,
                                                 const float* __restrict__ b,
                                                 const float* __restrict__ base,
                                                 float* __restrict__ outF,
                                                 __bf16* __restrict__ outB) {
  int row = blockIdx.x, tid = threadIdx.x;
  const float4 v = reinterpret_cast<const float4*>(x + (size_t)row * C_)[tid];
  float s  = v.x + v.y + v.z + v.w;
  float ss = v.x*v.x + v.y*v.y + v.z*v.z + v.w*v.w;
  #pragma unroll
  for (int m = 16; m >= 1; m >>= 1) { s += __shfl_xor(s, m); ss += __shfl_xor(ss, m); }
  __shared__ float sh[8];
  int wid = tid >> 5;
  if ((tid & 31) == 0) { sh[wid] = s; sh[wid + 4] = ss; }
  __syncthreads();
  float S  = sh[0] + sh[1] + sh[2] + sh[3];
  float SS = sh[4] + sh[5] + sh[6] + sh[7];
  float mean = S * (1.0f / C_);
  float var  = SS * (1.0f / C_) - mean * mean;
  float rstd = rsqrtf(var + EPS_);
  const float4 gg = reinterpret_cast<const float4*>(g)[tid];
  const float4 bb = reinterpret_cast<const float4*>(b)[tid];
  float y0 = (v.x - mean) * rstd * gg.x + bb.x;
  float y1 = (v.y - mean) * rstd * gg.y + bb.y;
  float y2 = (v.z - mean) * rstd * gg.z + bb.z;
  float y3 = (v.w - mean) * rstd * gg.w + bb.w;
  if (MODE == 0) {
    const float4 bs = reinterpret_cast<const float4*>(base + (size_t)row * C_)[tid];
    reinterpret_cast<float4*>(outF + (size_t)row * C_)[tid] =
        make_float4(bs.x + y0, bs.y + y1, bs.z + y2, bs.w + y3);
  } else {
    __bf16* o = outB + (size_t)row * C_ + tid * 4;
    o[0] = (__bf16)y0; o[1] = (__bf16)y1; o[2] = (__bf16)y2; o[3] = (__bf16)y3;
  }
}

// ---------------- CPE: h = sum_27 gather(feat) @ cpe_w[k]^T + cpe_b ----------------
__global__ __launch_bounds__(256) void cpe_conv_kernel(const __bf16* __restrict__ featb,
                                                       const __bf16* __restrict__ cpewb,
                                                       const float* __restrict__ cpe_b,
                                                       const int* __restrict__ nbr,
                                                       float* __restrict__ out) {
  __shared__ __align__(16) __bf16 stage[16][520];   // 16-row tile, padded rows
  const int tid  = threadIdx.x;
  const int lane = tid & 31, w = tid >> 5;
  const int n0 = blockIdx.x * 16;
  const int m  = lane & 15;
  const int hofA = (lane & 16) ? 16 : 0;
  const int bofB = (lane & 16) ? 16 : 0;
  const int colbase = w * 64;

  v8f acc[4] = {};
  const int srow = tid >> 4;           // staging: row 0..15
  const int scol = (tid & 15) * 32;    // staging: 32 bf16 per thread

  for (int kk = 0; kk < 27; ++kk) {
    // ---- stage gathered (masked) rows into LDS ----
    {
      int src = nbr[(n0 + srow) * 27 + kk];
      uint4* dst = reinterpret_cast<uint4*>(&stage[srow][scol]);
      if (src >= 0) {
        const uint4* sp = reinterpret_cast<const uint4*>(featb + (size_t)src * C_ + scol);
        dst[0] = sp[0]; dst[1] = sp[1]; dst[2] = sp[2]; dst[3] = sp[3];
      } else {
        uint4 z = make_uint4(0, 0, 0, 0);
        dst[0] = z; dst[1] = z; dst[2] = z; dst[3] = z;
      }
    }
    __syncthreads();
    const __bf16* wk = cpewb + (size_t)kk * C_ * C_;
    #pragma unroll 4
    for (int ks = 0; ks < 16; ++ks) {
      Frag16 a;
      const __bf16* ap = &stage[m][ks * 32 + hofA];
      a.q[0] = *reinterpret_cast<const uint4*>(ap);
      a.q[1] = *reinterpret_cast<const uint4*>(ap + 8);
      #pragma unroll
      for (int t = 0; t < 4; ++t) {
        int n = colbase + t * 16 + m;
        const __bf16* bp = wk + (size_t)n * C_ + ks * 32 + bofB;
        Frag16 bfr;
        bfr.q[0] = ld16g(bp);
        bfr.q[1] = ld16g(bp + 8);
        acc[t] = wmma_bf16(a.v, bfr.v, acc[t]);
      }
    }
    __syncthreads();
  }
  // ---- epilogue: + cpe_b, store f32 ----
  const int mh = (lane & 16) ? 8 : 0;
  #pragma unroll
  for (int t = 0; t < 4; ++t) {
    int col = colbase + t * 16 + m;
    float bias = cpe_b[col];
    #pragma unroll
    for (int r = 0; r < 8; ++r) {
      int row = n0 + r + mh;
      out[(size_t)row * C_ + col] = acc[t][r] + bias;
    }
  }
}

// ---------------- generic GEMM: out = A(bf16)[M,Cin] @ W(bf16)[Cout,Cin]^T + bias ----
// MODE 0: f32 store; MODE 1: gelu -> bf16 store; MODE 2: row-gather by rowidx, qkv split store
template<int MODE>
__global__ __launch_bounds__(256) void gemm_kernel(const __bf16* __restrict__ A,
                                                   const __bf16* __restrict__ W,
                                                   const float* __restrict__ bias,
                                                   const int* __restrict__ rowidx,
                                                   float* __restrict__ outF,
                                                   __bf16* __restrict__ outB,
                                                   __bf16* __restrict__ qb,
                                                   __bf16* __restrict__ kb,
                                                   __bf16* __restrict__ vT,
                                                   int Cin, int Cout) {
  const int lane = threadIdx.x & 31, w = threadIdx.x >> 5;
  const int row0 = blockIdx.x * 16;
  const int colbase = blockIdx.y * 512 + w * 64;
  const int m = lane & 15;
  const int hofA = (lane & 16) ? 16 : 0;
  const int bofB = (lane & 16) ? 16 : 0;

  int arow = row0 + m;
  if (MODE == 2) arow = rowidx[arow];
  const __bf16* Arow = A + (size_t)arow * Cin;

  v8f acc[4] = {};
  const int nsteps = Cin >> 5;
  for (int ks = 0; ks < nsteps; ++ks) {
    Frag16 a;
    const __bf16* ap = Arow + ks * 32 + hofA;
    a.q[0] = ld16g(ap);
    a.q[1] = ld16g(ap + 8);
    #pragma unroll
    for (int t = 0; t < 4; ++t) {
      int n = colbase + t * 16 + m;
      const __bf16* bp = W + (size_t)n * Cin + ks * 32 + bofB;
      Frag16 bfr;
      bfr.q[0] = ld16g(bp);
      bfr.q[1] = ld16g(bp + 8);
      acc[t] = wmma_bf16(a.v, bfr.v, acc[t]);
    }
  }
  const int mh = (lane & 16) ? 8 : 0;
  #pragma unroll
  for (int t = 0; t < 4; ++t) {
    int col = colbase + t * 16 + m;
    float bv = bias[col];
    #pragma unroll
    for (int r = 0; r < 8; ++r) {
      int row = row0 + r + mh;
      float v = acc[t][r] + bv;
      if (MODE == 0) {
        outF[(size_t)row * Cout + col] = v;
      } else if (MODE == 1) {
        float gl = 0.5f * v * (1.0f + erff(v * 0.70710678118f));
        outB[(size_t)row * Cout + col] = (__bf16)gl;
      } else {
        int which = col >> 9, hh = (col >> 6) & 7, d = col & 63;
        int p = row >> 10, tt = row & 1023;
        __bf16 bvv = (__bf16)v;
        if (which == 0)      qb[(size_t)((p * 8 + hh) * 1024 + tt) * 64 + d] = bvv;
        else if (which == 1) kb[(size_t)((p * 8 + hh) * 1024 + tt) * 64 + d] = bvv;
        else                 vT[(size_t)((p * 8 + hh) * 64 + d) * 1024 + tt] = bvv;
      }
    }
  }
}

// ---------------- flash attention: one wave per (p, h, 16-query tile) ----------------
__global__ __launch_bounds__(32) void attn_kernel(const __bf16* __restrict__ qb,
                                                  const __bf16* __restrict__ kb,
                                                  const __bf16* __restrict__ vT,
                                                  float* __restrict__ oserial) {
  __shared__ __align__(16) __bf16 pst[16 * 32];
  const int lane = threadIdx.x;
  const int qt = blockIdx.x, h = blockIdx.y, p = blockIdx.z;
  const int m = lane & 15;
  const int hofA = (lane & 16) ? 16 : 0;
  const int bofB = (lane & 16) ? 16 : 0;
  const int mh   = (lane & 16) ? 8  : 0;
  const size_t base = (size_t)(p * 8 + h) * 1024 * 64;

  // preload Q fragments (16 rows x 64 = 2 K-steps)
  Frag16 aq[2];
  const int trow = qt * 16 + m;
  #pragma unroll
  for (int ks = 0; ks < 2; ++ks) {
    const __bf16* qp = qb + base + (size_t)trow * 64 + ks * 32 + hofA;
    aq[ks].q[0] = ld16g(qp);
    aq[ks].q[1] = ld16g(qp + 8);
  }

  float mrow[8], lrow[8];
  #pragma unroll
  for (int r = 0; r < 8; ++r) { mrow[r] = -1e30f; lrow[r] = 0.0f; }
  v8f o[4] = {};

  for (int kc = 0; kc < 32; ++kc) {        // 32 key-chunks of 32
    v8f s0 = {}, s1 = {};
    #pragma unroll
    for (int ks = 0; ks < 2; ++ks) {
      const __bf16* kp0 = kb + base + (size_t)(kc * 32 + m) * 64 + ks * 32 + bofB;
      Frag16 b0, b1;
      b0.q[0] = ld16g(kp0);        b0.q[1] = ld16g(kp0 + 8);
      const __bf16* kp1 = kp0 + 16 * 64;
      b1.q[0] = ld16g(kp1);        b1.q[1] = ld16g(kp1 + 8);
      s0 = wmma_bf16(aq[ks].v, b0.v, s0);
      s1 = wmma_bf16(aq[ks].v, b1.v, s1);
    }
    float sc[8];
    #pragma unroll
    for (int r = 0; r < 8; ++r) {
      float a0 = s0[r] * SCALE_, a1 = s1[r] * SCALE_;
      float red = fmaxf(a0, a1);
      red = fmaxf(red, __shfl_xor(red, 1));
      red = fmaxf(red, __shfl_xor(red, 2));
      red = fmaxf(red, __shfl_xor(red, 4));
      red = fmaxf(red, __shfl_xor(red, 8));
      float mn = fmaxf(mrow[r], red);
      sc[r] = __expf(mrow[r] - mn);
      float p0 = __expf(a0 - mn), p1 = __expf(a1 - mn);
      pst[(r + mh) * 32 + m]      = (__bf16)p0;
      pst[(r + mh) * 32 + m + 16] = (__bf16)p1;
      float rs = p0 + p1;
      rs += __shfl_xor(rs, 1);
      rs += __shfl_xor(rs, 2);
      rs += __shfl_xor(rs, 4);
      rs += __shfl_xor(rs, 8);
      lrow[r] = lrow[r] * sc[r] + rs;
      mrow[r] = mn;
    }
    #pragma unroll
    for (int t = 0; t < 4; ++t)
      #pragma unroll
      for (int r = 0; r < 8; ++r) o[t][r] *= sc[r];
    __syncthreads();
    // P tile as A-fragment (16x32) from LDS
    Frag16 pa;
    const __bf16* pp = &pst[m * 32 + hofA];
    pa.q[0] = *reinterpret_cast<const uint4*>(pp);
    pa.q[1] = *reinterpret_cast<const uint4*>(pp + 8);
    #pragma unroll
    for (int t = 0; t < 4; ++t) {
      int d = t * 16 + m;
      const __bf16* vp = vT + (size_t)((p * 8 + h) * 64 + d) * 1024 + kc * 32 + bofB;
      Frag16 bv;
      bv.q[0] = ld16g(vp);
      bv.q[1] = ld16g(vp + 8);
      o[t] = wmma_bf16(pa.v, bv.v, o[t]);
    }
    __syncthreads();
  }
  // epilogue: divide by row sums, store f32 serialized
  #pragma unroll
  for (int r = 0; r < 8; ++r) {
    float inv = 1.0f / lrow[r];
    int row = p * 1024 + qt * 16 + r + mh;
    #pragma unroll
    for (int t = 0; t < 4; ++t) {
      int col = h * 64 + t * 16 + m;
      oserial[(size_t)row * C_ + col] = o[t][r] * inv;
    }
  }
}

// ---------------- host launch ----------------
extern "C" void kernel_launch(void* const* d_in, const int* in_sizes, int n_in,
                              void* d_out, int out_size, void* d_ws, size_t ws_size,
                              hipStream_t stream) {
  (void)in_sizes; (void)n_in; (void)out_size; (void)ws_size;
  char* ws = (char*)d_ws;
  auto bfp = [&](size_t off) { return (__bf16*)(ws + off); };

  const float* feat      = (const float*)d_in[0];
  const int*   nbr       = (const int*)  d_in[1];
  const int*   order     = (const int*)  d_in[2];
  const float* cpe_w     = (const float*)d_in[3];
  const float* cpe_b     = (const float*)d_in[4];
  const float* cpe_lin_w = (const float*)d_in[5];
  const float* cpe_lin_b = (const float*)d_in[6];
  const float* cpe_ln_g  = (const float*)d_in[7];
  const float* cpe_ln_b  = (const float*)d_in[8];
  const float* ln1_g     = (const float*)d_in[9];
  const float* ln1_b     = (const float*)d_in[10];
  const float* qkv_w     = (const float*)d_in[11];
  const float* qkv_b     = (const float*)d_in[12];
  const float* proj_w    = (const float*)d_in[13];
  const float* proj_b    = (const float*)d_in[14];
  const float* ln2_g     = (const float*)d_in[15];
  const float* ln2_b     = (const float*)d_in[16];
  const float* fc1_w     = (const float*)d_in[17];
  const float* fc1_b     = (const float*)d_in[18];
  const float* fc2_w     = (const float*)d_in[19];
  const float* fc2_b     = (const float*)d_in[20];

  float*  tmpF  = (float*)(ws + OFF_TMPF);
  float*  featR = (float*)(ws + OFF_FEATR);
  __bf16* act1  = bfp(OFF_ACT1);
  __bf16* qb    = bfp(OFF_ACT4);
  __bf16* kb    = qb + (size_t)N_ * C_;
  __bf16* vT    = kb + (size_t)N_ * C_;
  float*  outF  = (float*)d_out;

  auto cvt = [&](const float* src, __bf16* dst, int n) {
    f2bf_kernel<<<(n + 255) / 256, 256, 0, stream>>>(src, dst, n);
  };

  // 0) weights + input activations -> bf16
  cvt(cpe_w,     bfp(OFF_CPEW),   27 * 512 * 512);
  cvt(cpe_lin_w, bfp(OFF_CPELIN), 512 * 512);
  cvt(qkv_w,     bfp(OFF_QKVW),   3 * 512 * 512);
  cvt(proj_w,    bfp(OFF_PROJW),  512 * 512);
  cvt(fc1_w,     bfp(OFF_FC1W),   2048 * 512);
  cvt(fc2_w,     bfp(OFF_FC2W),   2048 * 512);
  cvt(feat,      act1,            N_ * C_);

  // 1) CPE conv (27 gathered matmuls) -> tmpF
  cpe_conv_kernel<<<N_ / 16, 256, 0, stream>>>(act1, bfp(OFF_CPEW), cpe_b, nbr, tmpF);
  // 2) cpe_lin: bf16(tmpF) @ cpe_lin_w^T -> tmpF
  cvt(tmpF, act1, N_ * C_);
  gemm_kernel<0><<<dim3(N_ / 16, 1), 256, 0, stream>>>(act1, bfp(OFF_CPELIN), cpe_lin_b,
      nullptr, tmpF, nullptr, nullptr, nullptr, nullptr, 512, 512);
  // 3) featR = feat + LN(tmpF)
  ln_kernel<0><<<N_, 128, 0, stream>>>(tmpF, cpe_ln_g, cpe_ln_b, feat, featR, nullptr);
  // 4) x1 = bf16(LN(featR, ln1))
  ln_kernel<1><<<N_, 128, 0, stream>>>(featR, ln1_g, ln1_b, nullptr, nullptr, act1);
  // 5) qkv gemm with row-gather by `order`, split-store q/k/vT bf16
  gemm_kernel<2><<<dim3(N_ / 16, 3), 256, 0, stream>>>(act1, bfp(OFF_QKVW), qkv_b,
      order, nullptr, nullptr, qb, kb, vT, 512, 1536);
  // 6) attention -> o_serial (tmpF)
  attn_kernel<<<dim3(K_ / 16, H_, P_), 32, 0, stream>>>(qb, kb, vT, tmpF);
  // 7) scatter back to original order as bf16
  scatter_kernel<<<(N_ * C_) / 256, 256, 0, stream>>>(tmpF, order, act1);
  // 8) proj -> tmpF ; featR += tmpF
  gemm_kernel<0><<<dim3(N_ / 16, 1), 256, 0, stream>>>(act1, bfp(OFF_PROJW), proj_b,
      nullptr, tmpF, nullptr, nullptr, nullptr, nullptr, 512, 512);
  add_kernel<<<(N_ * C_ / 4 + 255) / 256, 256, 0, stream>>>(
      (const float4*)featR, (const float4*)tmpF, (float4*)featR, N_ * C_ / 4);
  // 9) MLP: x2 = bf16(LN(featR, ln2)); h = gelu(x2@fc1^T) bf16; tmpF = h@fc2^T
  ln_kernel<1><<<N_, 128, 0, stream>>>(featR, ln2_g, ln2_b, nullptr, nullptr, act1);
  gemm_kernel<1><<<dim3(N_ / 16, 4), 256, 0, stream>>>(act1, bfp(OFF_FC1W), fc1_b,
      nullptr, nullptr, qb /* = act4 region */, nullptr, nullptr, nullptr, 512, 2048);
  gemm_kernel<0><<<dim3(N_ / 16, 1), 256, 0, stream>>>(qb, bfp(OFF_FC2W), fc2_b,
      nullptr, tmpF, nullptr, nullptr, nullptr, nullptr, 2048, 512);
  // 10) out = featR + tmpF
  add_kernel<<<(N_ * C_ / 4 + 255) / 256, 256, 0, stream>>>(
      (const float4*)featR, (const float4*)tmpF, (float4*)outF, N_ * C_ / 4);
}
